// Attention_Rel_Scl_74131135529191
// MI455X (gfx1250) — matrix operand order repaired and checked by reading, not compiled, not run on hardware
//
#include <hip/hip_runtime.h>
#include <hip/hip_bf16.h>

// ---------------------------------------------------------------------------
// Gated attention (Attention_Rel_Scl) for MI455X / gfx1250, wave32 + WMMA.
//   B=8, S=1024, E=512, H=8, d=64.
// Pass 1: Q/K/V projections via v_wmma_f32_16x16x32_f16 (f32->f16 inline),
//         V stored transposed so attn@V B-fragments are contiguous.
// Pass 2: fused scores->softmax->gate blend->attn write->attn@V, with the
//         32x1024 score block and mask block resident in LDS (258KB/WGP).
//         Mask is streamed HBM->LDS with GLOBAL_LOAD_ASYNC_TO_LDS_B128
//         (ASYNCcnt-tracked) overlapped with the score WMMAs.
// ---------------------------------------------------------------------------

typedef __attribute__((ext_vector_type(16))) _Float16 v16h;
typedef __attribute__((ext_vector_type(8)))  _Float16 v8h;
typedef __attribute__((ext_vector_type(8)))  float    v8f;
typedef __attribute__((ext_vector_type(4)))  float    v4f;
typedef __attribute__((ext_vector_type(4)))  int      v4i;

#define EMB   512
#define HEADS 8
#define BATCH 8
#define SEQ   1024
#define HD    64
#define BH    (BATCH * HEADS)
#define ROWS  32          // query rows per workgroup in pass 2

#if defined(__has_builtin)
#if __has_builtin(__builtin_amdgcn_global_load_async_to_lds_b128)
#define USE_ASYNC_LDS 1
#endif
#endif

#if USE_ASYNC_LDS
typedef __attribute__((address_space(1))) v4i gv4i;   // global int4
typedef __attribute__((address_space(3))) v4i sv4i;   // LDS int4
#endif

union H16 { v8h h[2]; v16h v; };

__device__ inline v8f wmma_f16(v16h a, v16h b, v8f c) {
    // D = A(16x32 f16) * B(32x16 f16) + C(16x16 f32)
    return __builtin_amdgcn_wmma_f32_16x16x32_f16(
        /*neg_a=*/false, a, /*neg_b=*/false, b,
        /*c_mod=*/(short)0, c, /*reuse_a=*/false, /*reuse_b=*/false);
}

// 16 contiguous f16 -> one lane's B fragment slice (contiguous k order used here: B[k0+(L/16)*16+i][L%16])
__device__ inline v16h load16h(const _Float16* p) {
    H16 u;
    u.h[0] = *reinterpret_cast<const v8h*>(p);
    u.h[1] = *reinterpret_cast<const v8h*>(p + 8);
    return u.v;
}

// A fragment (16x32, f16 row-major source). Lane layout: K = {base..base+15},
// base = (lane/16)*16 (contiguous, matching B fragments).
__device__ inline v16h loadA_f16(const _Float16* rowp, int k0, int lane) {
    const int base = (lane >> 4) * 16;
    H16 u;
    u.h[0] = *reinterpret_cast<const v8h*>(rowp + k0 + base);
    u.h[1] = *reinterpret_cast<const v8h*>(rowp + k0 + base + 8);
    return u.v;
}

// A fragment built from an f32 row (convert inline to f16)
__device__ inline v16h loadA_f32(const float* rowp, int k0, int lane) {
    const int base = (lane >> 4) * 16;
    v16h a;
#pragma unroll
    for (int i = 0; i < 8; ++i) a[i]     = (_Float16)rowp[k0 + base + i];
#pragma unroll
    for (int i = 0; i < 8; ++i) a[8 + i] = (_Float16)rowp[k0 + base + 8 + i];
    return a;
}

// B fragment from f32 row-major W (row = output feature, 16 contiguous f32)
__device__ inline v16h loadB_f32(const float* rowp, int k0, int lane) {
    const float* p = rowp + k0 + (lane >> 4) * 16;
    v16h b;
#pragma unroll
    for (int i = 0; i < 16; ++i) b[i] = (_Float16)p[i];
    return b;
}

// ---------------------------------------------------------------------------
// Pass 1: Y = X @ W^T for z = 0(Q) / 1(K) / 2(V).  M=8192, N=512, Kc=512.
// 256 threads = 8 waves: wave = (msub 0..3) x (nsub 0..1), tile 64x64/block.
// Q,K stored f16 [bh][s][d]; V stored f16 transposed [bh][d][s].
// ---------------------------------------------------------------------------
__global__ __launch_bounds__(256)
void proj_kernel(const float* __restrict__ q_in, const float* __restrict__ k_in,
                 const float* __restrict__ v_in,
                 const float* __restrict__ Wq, const float* __restrict__ Wk,
                 const float* __restrict__ Wv,
                 _Float16* __restrict__ Q16, _Float16* __restrict__ K16,
                 _Float16* __restrict__ Vt16) {
    const int z = blockIdx.z;
    const float* X = (z == 0) ? q_in : (z == 1) ? k_in : v_in;
    const float* W = (z == 0) ? Wq   : (z == 1) ? Wk   : Wv;

    const int tid  = threadIdx.x;
    const int lane = tid & 31;
    const int wave = tid >> 5;
    const int msub = wave & 3;
    const int nsub = wave >> 2;

    const int m0 = blockIdx.x * 64 + msub * 16;
    const int n0 = blockIdx.y * 64 + nsub * 32;

    const float* xrow = X + (m0 + (lane & 15)) * EMB;
    const float* w0   = W + (n0 +      (lane & 15)) * EMB;
    const float* w1   = W + (n0 + 16 + (lane & 15)) * EMB;

    v8f c0 = {}; v8f c1 = {};
    for (int k0 = 0; k0 < EMB; k0 += 32) {
        v16h a  = loadA_f32(xrow, k0, lane);
        v16h b0 = loadB_f32(w0, k0, lane);
        c0 = wmma_f16(a, b0, c0);
        v16h b1 = loadB_f32(w1, k0, lane);
        c1 = wmma_f16(a, b1, c1);
    }

    const int mbase = m0 + 8 * (lane >> 4);
#pragma unroll
    for (int nt = 0; nt < 2; ++nt) {
        v8f c = nt ? c1 : c0;
        const int gn = n0 + nt * 16 + (lane & 15);
        const int h = gn >> 6, d = gn & 63;
#pragma unroll
        for (int r = 0; r < 8; ++r) {
            const int gm = mbase + r;
            const int b = gm >> 10, s = gm & 1023;
            const int bh = b * HEADS + h;
            const _Float16 val = (_Float16)c[r];
            if (z == 0)      Q16 [bh * (SEQ * HD) + s * HD + d] = val;
            else if (z == 1) K16 [bh * (SEQ * HD) + s * HD + d] = val;
            else             Vt16[bh * (SEQ * HD) + d * SEQ + s] = val;
        }
    }
}

// ---------------------------------------------------------------------------
// Pass 2: fused attention for one (bh, 32-query block).
// LDS: sc[32][1024] f32 (scores -> content exp -> attn), mk[32][1024] f32
// (mask -> pos exp), red/red2[32][8] partials.  258KB <= 320KB/WGP.
// ---------------------------------------------------------------------------
__global__ __launch_bounds__(256)
void attn_kernel(const _Float16* __restrict__ Q16, const _Float16* __restrict__ K16,
                 const _Float16* __restrict__ Vt16, const float* __restrict__ mask,
                 const float* __restrict__ gating, float* __restrict__ out,
                 float* __restrict__ attn_out) {
    extern __shared__ float smem[];
    float* sc   = smem;                       // [ROWS][SEQ]
    float* mk   = smem + ROWS * SEQ;          // [ROWS][SEQ]
    float* red  = smem + 2 * ROWS * SEQ;      // [ROWS][8]
    float* red2 = red + ROWS * 8;             // [ROWS][8]

    const int bh = blockIdx.x;                // 0..63
    const int q0 = blockIdx.y * ROWS;
    const int h  = bh & (HEADS - 1);

    const int tid  = threadIdx.x;
    const int lane = tid & 31;
    const int wave = tid >> 5;

    const _Float16* Qb = Q16  + bh * (SEQ * HD);
    const _Float16* Kb = K16  + bh * (SEQ * HD);
    const _Float16* Vb = Vt16 + bh * (SEQ * HD);

    // ---- stream the mask block into LDS (read once from HBM) ----
    // Issued BEFORE the score WMMAs so the async DMA overlaps matrix work.
    {
        const int mrow = tid >> 3, mcb = (tid & 7) * 128;
        const float* mp = mask + ((size_t)bh * SEQ + (q0 + mrow)) * SEQ + mcb;
        float* dst = mk + mrow * SEQ + mcb;
#if USE_ASYNC_LDS
        // per-lane 16B HBM -> LDS transfers, no VGPR staging (ASYNCcnt-tracked)
        for (int j = 0; j < 128; j += 4) {
            gv4i* gp = (gv4i*)(uintptr_t)(mp + j);
            sv4i* lp = (sv4i*)(unsigned int)(uintptr_t)(dst + j);
            __builtin_amdgcn_global_load_async_to_lds_b128(gp, lp, 0, 0);
        }
#else
        for (int j = 0; j < 128; j += 4)
            *reinterpret_cast<v4f*>(dst + j) = *reinterpret_cast<const v4f*>(mp + j);
#endif
    }

    // ---- scores = scale * Q Kt  (2 m-tiles x 4 n-chunks over 8 waves) ----
    {
        const int msub  = wave & 1;
        const int nbase = (wave >> 1) * 256;
        const _Float16* qrow = Qb + (q0 + msub * 16 + (lane & 15)) * HD;
        const v16h a0 = loadA_f16(qrow, 0,  lane);
        const v16h a1 = loadA_f16(qrow, 32, lane);
        const float scale = 0.04419417382415922f;   // 512^-0.5
        const int lrow = msub * 16 + 8 * (lane >> 4);
        for (int nt = 0; nt < 16; ++nt) {
            const int n = nbase + nt * 16;
            const _Float16* kp = Kb + (n + (lane & 15)) * HD + (lane >> 4) * 16;
            v8f c = {};
            c = wmma_f16(a0, load16h(kp),      c);   // kc = 0..31
            c = wmma_f16(a1, load16h(kp + 32), c);   // kc = 32..63
            const int col = n + (lane & 15);
#pragma unroll
            for (int r = 0; r < 8; ++r)
                sc[(lrow + r) * SEQ + col] = c[r] * scale;
        }
    }

#if USE_ASYNC_LDS
    // drain this wave's async HBM->LDS transfers before the workgroup barrier
#if __has_builtin(__builtin_amdgcn_s_wait_asynccnt)
    __builtin_amdgcn_s_wait_asynccnt(0);
#else
    asm volatile("s_wait_asynccnt 0" ::: "memory");
#endif
#endif
    __syncthreads();

    // ---- dual softmax + gated blend + renormalize (8 threads per row) ----
    const int row = tid >> 3;
    const int cb  = (tid & 7) * 128;
    float* scr = sc + row * SEQ;
    float* mkr = mk + row * SEQ;
    const int rs = row * 8 + (tid & 7);

    float cmax = -3.402823e38f, mmax = -3.402823e38f;
    for (int j = 0; j < 128; ++j) {
        cmax = fmaxf(cmax, scr[cb + j]);
        mmax = fmaxf(mmax, mkr[cb + j]);
    }
    red[rs] = cmax; red2[rs] = mmax;
    __syncthreads();
    float rcm = red[row * 8], rmm = red2[row * 8];
#pragma unroll
    for (int j = 1; j < 8; ++j) {
        rcm = fmaxf(rcm, red[row * 8 + j]);
        rmm = fmaxf(rmm, red2[row * 8 + j]);
    }
    __syncthreads();

    float csum = 0.f, msum = 0.f;
    for (int j = 0; j < 128; ++j) {
        float e1 = __expf(scr[cb + j] - rcm); scr[cb + j] = e1; csum += e1;
        float e2 = __expf(mkr[cb + j] - rmm); mkr[cb + j] = e2; msum += e2;
    }
    red[rs] = csum; red2[rs] = msum;
    __syncthreads();
    float rcs = 0.f, rms = 0.f;
#pragma unroll
    for (int j = 0; j < 8; ++j) { rcs += red[row * 8 + j]; rms += red2[row * 8 + j]; }
    __syncthreads();

    const float g  = 1.f / (1.f + __expf(-gating[h]));
    const float w1 = (1.f - g) / rcs;
    const float w2 = g / rms;
    float tsum = 0.f;
    for (int j = 0; j < 128; ++j) {
        float v = w1 * scr[cb + j] + w2 * mkr[cb + j];
        scr[cb + j] = v; tsum += v;
    }
    red[rs] = tsum;
    __syncthreads();
    float rts = 0.f;
#pragma unroll
    for (int j = 0; j < 8; ++j) rts += red[row * 8 + j];

    const float inv = 1.f / rts;
    float* gattn = attn_out + ((size_t)bh * SEQ + (q0 + row)) * SEQ + cb;
    for (int j = 0; j < 128; j += 4) {
        v4f v = *reinterpret_cast<v4f*>(scr + cb + j);
        v = v * inv;
        *reinterpret_cast<v4f*>(scr + cb + j) = v;            // keep for attn@V
        *reinterpret_cast<v4f*>(gattn + j)    = v;            // required output
    }
    __syncthreads();

    // ---- out = attn @ V  (2 m-tiles x 4 d-tiles over 8 waves, Kc=1024) ----
    {
        const int msub  = wave & 1;
        const int ntile = wave >> 1;                  // 0..3 (d chunks of 16)
        const float* arow = sc + (msub * 16 + (lane & 15)) * SEQ;
        const int base = (lane >> 4) * 16;
        const _Float16* vbase = Vb + (ntile * 16 + (lane & 15)) * SEQ + (lane >> 4) * 16;
        v8f c = {};
        for (int k0 = 0; k0 < SEQ; k0 += 32) {
            v16h a;
#pragma unroll
            for (int i = 0; i < 8; ++i) a[i]     = (_Float16)arow[k0 + base + i];
#pragma unroll
            for (int i = 0; i < 8; ++i) a[8 + i] = (_Float16)arow[k0 + base + 8 + i];
            c = wmma_f16(a, load16h(vbase + k0), c);
        }
        const int e    = h * HD + ntile * 16 + (lane & 15);
        const int b_   = bh >> 3;
        const int srow = q0 + msub * 16 + 8 * (lane >> 4);
#pragma unroll
        for (int r = 0; r < 8; ++r)
            out[((size_t)(b_ * SEQ + srow + r)) * EMB + e] = c[r];
    }
}

// ---------------------------------------------------------------------------
extern "C" void kernel_launch(void* const* d_in, const int* in_sizes, int n_in,
                              void* d_out, int out_size, void* d_ws, size_t ws_size,
                              hipStream_t stream) {
    const float* q    = (const float*)d_in[0];
    const float* k    = (const float*)d_in[1];
    const float* v    = (const float*)d_in[2];
    const float* mask = (const float*)d_in[3];
    const float* Wq   = (const float*)d_in[4];
    const float* Wk   = (const float*)d_in[5];
    const float* Wv   = (const float*)d_in[6];
    const float* gate = (const float*)d_in[7];

    float* out  = (float*)d_out;                       // [B,S,E]
    float* attn = out + (size_t)BATCH * SEQ * EMB;     // [B*H,S,S]

    _Float16* Q16  = (_Float16*)d_ws;                  // 8 MB
    _Float16* K16  = Q16 + (size_t)BH * SEQ * HD;      // 8 MB
    _Float16* Vt16 = K16 + (size_t)BH * SEQ * HD;      // 8 MB (transposed)

    dim3 g1((BATCH * SEQ) / 64, EMB / 64, 3);
    proj_kernel<<<g1, 256, 0, stream>>>(q, k, v, Wq, Wk, Wv, Q16, K16, Vt16);

    const size_t smem = (size_t)(2 * ROWS * SEQ + 2 * ROWS * 8) * sizeof(float);
    (void)hipFuncSetAttribute(reinterpret_cast<const void*>(attn_kernel),
                              hipFuncAttributeMaxDynamicSharedMemorySize, (int)smem);
    dim3 g2(BH, SEQ / ROWS);
    attn_kernel<<<g2, 256, smem, stream>>>(Q16, K16, Vt16, mask, gate, out, attn);
}
